// MYLSTM_89429809038101
// MI455X (gfx1250) — matrix operand order repaired and checked by reading, not compiled, not run on hardware
//
#include <hip/hip_runtime.h>

// ---------------------------------------------------------------------------
// 2-layer LSTM (B=128, T=1024, D=H=512) for gfx1250 (MI455X).
// Persistent kernel, bf16 WMMA (v_wmma_f32_16x16x32_bf16), fp32 accum,
// weights pre-packed into WMMA B-operand lane layout, async global->LDS
// double-buffered weight staging, grid barriers via agent-scope atomics.
// ---------------------------------------------------------------------------

typedef __bf16 bf16;
typedef __attribute__((ext_vector_type(16))) __bf16 v16bf;
typedef __attribute__((ext_vector_type(8)))  float  v8f;
typedef __attribute__((ext_vector_type(4)))  int    v4i_;

// Address-space-qualified pointer types matching the async-load builtin's
// signature (v4i pointee in AS1 global / AS3 LDS).
typedef __attribute__((address_space(1))) v4i_* gas_v4i_ptr;
typedef __attribute__((address_space(3))) v4i_* las_v4i_ptr;

#define NBLK   32      // persistent workgroups (one per 16 h-columns)
#define TSTEPS 1024
#define HDIM   512

union BF16x16 {
  v16bf v;
  uint4 q[2];
};

#if defined(__AMDGCN__) && __has_builtin(__builtin_amdgcn_global_load_async_to_lds_b128)
#define HAVE_ASYNC 1
#else
#define HAVE_ASYNC 0
#endif

__device__ __forceinline__ void async_copy16(const void* gsrc, void* ldst) {
#if HAVE_ASYNC
  __builtin_amdgcn_global_load_async_to_lds_b128(
      (gas_v4i_ptr)(unsigned long long)gsrc,
      (las_v4i_ptr)(unsigned int)(unsigned long long)ldst,
      0, 0);
#else
  *(uint4*)ldst = *(const uint4*)gsrc;
#endif
}

__device__ __forceinline__ void wait_async_done() {
#if HAVE_ASYNC
#if __has_builtin(__builtin_amdgcn_s_wait_asynccnt)
  __builtin_amdgcn_s_wait_asynccnt(0);
#else
  asm volatile("s_wait_asynccnt 0" ::: "memory");
#endif
#endif
}

__device__ __forceinline__ float sigmoidf_(float x) {
  return 1.0f / (1.0f + __expf(-x));
}

// Stage one chunk (4 gates x 8 K-tiles x 1KB = 32KB) of pre-packed weights
// into LDS. Executed by wave 0 only; tracked with ASYNCcnt.
__device__ __forceinline__ void issue_chunk(const bf16* packW, int layer, int wg,
                                            int ch, unsigned char* ldsbuf, int lane) {
#pragma unroll
  for (int g = 0; g < 4; ++g) {
    const bf16* gbase = packW + ((((long)(layer * 4 + g)) * 32 + wg) * 32 + ch * 8) * 512;
    unsigned char* ldsg = ldsbuf + g * 8192;
#pragma unroll
    for (int j = 0; j < 16; ++j) {
      async_copy16(gbase + j * 256 + lane * 8, ldsg + j * 512 + lane * 16);
    }
  }
}

// One chunk of the gate GEMM: 8 K-tiles of 16x16x32 bf16 WMMA into 4 gate
// accumulators. A operand gathered per ISA A-layout:
//   lane<16 : row=lane,    K = {k0..k0+7, k0+16..k0+23}
//   lane>=16: row=lane-16, K = {k0+8..k0+15, k0+24..k0+31}
// B operand read from LDS in packed lane-major layout (2x ds_load_b128).
template <bool AF32>
__device__ __forceinline__ void compute_chunk(v8f (&acc)[4], const unsigned char* ldsbuf,
                                              const void* a_src, long a_stride,
                                              int khalf, int wrow, int lane) {
  const int row  = wrow + (lane & 15);
  const int ksel = (lane & 16) ? 8 : 0;
#pragma unroll
  for (int ktc = 0; ktc < 8; ++ktc) {
    const int k0 = (khalf * 8 + ktc) * 32 + ksel;
    BF16x16 A;
    if (AF32) {
      const float* p = (const float*)a_src + (long)row * a_stride + k0;
      float4 f0 = *(const float4*)(p);
      float4 f1 = *(const float4*)(p + 4);
      float4 f2 = *(const float4*)(p + 16);
      float4 f3 = *(const float4*)(p + 20);
      A.v[0]  = (bf16)f0.x; A.v[1]  = (bf16)f0.y; A.v[2]  = (bf16)f0.z; A.v[3]  = (bf16)f0.w;
      A.v[4]  = (bf16)f1.x; A.v[5]  = (bf16)f1.y; A.v[6]  = (bf16)f1.z; A.v[7]  = (bf16)f1.w;
      A.v[8]  = (bf16)f2.x; A.v[9]  = (bf16)f2.y; A.v[10] = (bf16)f2.z; A.v[11] = (bf16)f2.w;
      A.v[12] = (bf16)f3.x; A.v[13] = (bf16)f3.y; A.v[14] = (bf16)f3.z; A.v[15] = (bf16)f3.w;
    } else {
      const bf16* p = (const bf16*)a_src + (long)row * a_stride + k0;
      A.q[0] = *(const uint4*)(p);
      A.q[1] = *(const uint4*)(p + 16);
    }
#pragma unroll
    for (int g = 0; g < 4; ++g) {
      BF16x16 Bv;
      const unsigned char* bp = ldsbuf + (g * 8 + ktc) * 1024 + lane * 32;
      Bv.q[0] = *(const uint4*)(bp);
      Bv.q[1] = *(const uint4*)(bp + 16);
      acc[g] = __builtin_amdgcn_wmma_f32_16x16x32_bf16(false, A.v, false, Bv.v,
                                                       (short)0, acc[g], false, false);
    }
  }
}

// One LSTM cell step for this workgroup's 16 h-columns x this wave's 16 rows.
template <bool IH_F32>
__device__ __forceinline__ void lstm_cell(const void* a_ih, long ih_stride,
                                          const bf16* a_hh,
                                          const bf16* packW, int layer,
                                          const float* b_ih, const float* b_hh,
                                          unsigned char* smem,
                                          v8f& cstate, bf16* h_wr,
                                          float* out_h, float* out_h2, float* out_c,
                                          int wg, int wave, int lane, bool issuer) {
  const int hcol = wg * 16 + (lane & 15);
  const int wrow = wave * 16;
  v8f acc[4];
#pragma unroll
  for (int g = 0; g < 4; ++g) {
    const float b = b_ih[g * HDIM + hcol] + b_hh[g * HDIM + hcol];
#pragma unroll
    for (int e = 0; e < 8; ++e) acc[g][e] = b;
  }

  if (issuer) issue_chunk(packW, layer, wg, 0, smem, lane);
#pragma unroll 1
  for (int ch = 0; ch < 4; ++ch) {
    wait_async_done();            // chunk ch resident in LDS
    __syncthreads();
    if (issuer && ch < 3)         // prefetch next chunk into the other buffer
      issue_chunk(packW, layer, wg, ch + 1, smem + ((ch + 1) & 1) * 32768, lane);
    const unsigned char* buf = smem + (ch & 1) * 32768;
    if (ch < 2) {                 // x @ W_ih^T (or h0 @ W_ih1^T)
      if (IH_F32) compute_chunk<true >(acc, buf, a_ih, ih_stride, ch & 1, wrow, lane);
      else        compute_chunk<false>(acc, buf, a_ih, HDIM,      ch & 1, wrow, lane);
    } else {                      // h_prev @ W_hh^T
      compute_chunk<false>(acc, buf, a_hh, HDIM, ch & 1, wrow, lane);
    }
    __syncthreads();              // all reads of buf done before it is refilled
  }

  // activations + state update; accumulator element e <-> row (e + 8*(lane>=16))
#pragma unroll
  for (int e = 0; e < 8; ++e) {
    const float iv = sigmoidf_(acc[0][e]);
    const float fv = sigmoidf_(acc[1][e]);
    const float gv = tanhf(acc[2][e]);
    const float ov = sigmoidf_(acc[3][e]);
    const float cn = fv * cstate[e] + iv * gv;
    cstate[e] = cn;
    const float hv = ov * tanhf(cn);
    const int m = wrow + e + ((lane & 16) ? 8 : 0);
    const long idx = (long)m * HDIM + hcol;
    h_wr[idx] = (bf16)hv;
    if (out_h)  out_h[idx]  = hv;
    if (out_h2) out_h2[idx] = hv;
    if (out_c)  out_c[idx]  = cn;
  }
}

__device__ __forceinline__ void grid_barrier(unsigned* cnt, unsigned target) {
  __threadfence();
  __syncthreads();
  if (threadIdx.x == 0) {
    __hip_atomic_fetch_add(cnt, 1u, __ATOMIC_RELEASE, __HIP_MEMORY_SCOPE_AGENT);
    while (__hip_atomic_load(cnt, __ATOMIC_ACQUIRE, __HIP_MEMORY_SCOPE_AGENT) < target) {
      __builtin_amdgcn_s_sleep(2);
    }
  }
  __syncthreads();
  __threadfence();
}

// ---------------------------------------------------------------------------
// Prologue 1: zero the barrier counter and the bf16 h ping-pong buffers.
__global__ void init_ws_kernel(unsigned* cnt, unsigned short* h0, unsigned short* h1) {
  const long i = (long)blockIdx.x * blockDim.x + threadIdx.x;
  if (i == 0) *cnt = 0u;
  if (i < 131072) { h0[i] = 0; h1[i] = 0; }
}

// Prologue 2: pack W_{ih,hh}{0,1} (fp32 [2048,512]) into bf16 WMMA B-tiles.
// Tile id = ((layer*4+gate)*32 + colblk)*32 + kstream; 1KB per tile;
// lane n (<16) holds column n, K = kt*32+{0..7,16..23} ; lane n+16 holds K = +{8..15,24..31}.
__global__ void pack_weights_kernel(const float* __restrict__ Wih0,
                                    const float* __restrict__ Whh0,
                                    const float* __restrict__ Wih1,
                                    const float* __restrict__ Whh1,
                                    bf16* __restrict__ packW) {
  const int tid  = blockIdx.x * blockDim.x + threadIdx.x;
  const int lane = tid & 31;
  const int tile = tid >> 5;                // 8192 tiles
  const int kt = tile & 31;
  const int c  = (tile >> 5) & 31;
  const int g  = (tile >> 10) & 3;
  const int l  = (tile >> 12) & 1;
  const float* W = (l == 0) ? ((kt < 16) ? Wih0 : Whh0)
                            : ((kt < 16) ? Wih1 : Whh1);
  const int col = g * 512 + c * 16 + (lane & 15);
  const int k0  = (kt & 15) * 32 + ((lane & 16) ? 8 : 0);
  const float* src = W + (long)col * 512 + k0;
  bf16* dst = packW + (long)tile * 512 + lane * 16;
#pragma unroll
  for (int i = 0; i < 16; ++i) dst[i] = (bf16)src[i + ((i >= 8) ? 8 : 0)];
}

// ---------------------------------------------------------------------------
// Main persistent kernel: 32 workgroups x 256 threads, loops t=0..1023,
// two grid barriers per step (layer0 -> layer1 -> next t).
__global__ __launch_bounds__(256) void lstm2_persistent_kernel(
    const float* __restrict__ x,
    const float* __restrict__ b_ih0, const float* __restrict__ b_hh0,
    const float* __restrict__ b_ih1, const float* __restrict__ b_hh1,
    const bf16* __restrict__ packW,
    bf16* __restrict__ h0buf, bf16* __restrict__ h1buf,
    unsigned* __restrict__ cnt,
    float* __restrict__ out) {
  __shared__ __align__(16) unsigned char smem[65536];  // 2 x 32KB weight buffers
  const int wg   = blockIdx.x;
  const int tid  = threadIdx.x;
  const int wave = tid >> 5;
  const int lane = tid & 31;
  const bool issuer = (tid < 32);

  v8f c0, c1;
#pragma unroll
  for (int e = 0; e < 8; ++e) { c0[e] = 0.0f; c1[e] = 0.0f; }

  unsigned phase = 0;
#pragma unroll 1
  for (int t = 0; t < TSTEPS; ++t) {
    const int wr = t & 1, rd = wr ^ 1;
    const bool last = (t == TSTEPS - 1);
    const float* xt = x + (long)t * HDIM;   // row b lives at + b*(T*D)

    // layer 0: A_ih = x_t (fp32, converted on the fly), A_hh = h0(t-1) bf16
    lstm_cell<true>(xt, (long)TSTEPS * HDIM,
                    h0buf + (long)rd * 65536,
                    packW, 0, b_ih0, b_hh0, smem, c0,
                    h0buf + (long)wr * 65536,
                    last ? out + 65536  : nullptr,   // h stack layer0
                    nullptr,
                    last ? out + 196608 : nullptr,   // c stack layer0
                    wg, wave, lane, issuer);
    grid_barrier(cnt, (++phase) * NBLK);

    // layer 1: A_ih = h0(t) bf16, A_hh = h1(t-1) bf16
    lstm_cell<false>(h0buf + (long)wr * 65536, HDIM,
                     h1buf + (long)rd * 65536,
                     packW, 1, b_ih1, b_hh1, smem, c1,
                     h1buf + (long)wr * 65536,
                     last ? out + 131072 : nullptr,  // h stack layer1
                     last ? out          : nullptr,  // h[-1]
                     last ? out + 262144 : nullptr,  // c stack layer1
                     wg, wave, lane, issuer);
    grid_barrier(cnt, (++phase) * NBLK);
  }
}

// ---------------------------------------------------------------------------
extern "C" void kernel_launch(void* const* d_in, const int* in_sizes, int n_in,
                              void* d_out, int out_size, void* d_ws, size_t ws_size,
                              hipStream_t stream) {
  const float* x     = (const float*)d_in[0];
  const float* W_ih0 = (const float*)d_in[1];
  const float* b_ih0 = (const float*)d_in[2];
  const float* W_hh0 = (const float*)d_in[3];
  const float* b_hh0 = (const float*)d_in[4];
  const float* W_ih1 = (const float*)d_in[5];
  const float* b_ih1 = (const float*)d_in[6];
  const float* W_hh1 = (const float*)d_in[7];
  const float* b_hh1 = (const float*)d_in[8];
  float* out = (float*)d_out;

  char* ws = (char*)d_ws;
  unsigned* cnt = (unsigned*)ws;                         // barrier counter
  bf16* h0buf = (bf16*)(ws + 4096);                      // 2 x [128,512] bf16
  bf16* h1buf = (bf16*)(ws + 4096 + 262144);             // 2 x [128,512] bf16
  bf16* packW = (bf16*)(ws + 4096 + 2 * 262144);         // 8 MB packed weights

  init_ws_kernel<<<512, 256, 0, stream>>>(cnt, (unsigned short*)h0buf,
                                          (unsigned short*)h1buf);
  pack_weights_kernel<<<1024, 256, 0, stream>>>(W_ih0, W_hh0, W_ih1, W_hh1, packW);
  lstm2_persistent_kernel<<<NBLK, 256, 0, stream>>>(x, b_ih0, b_hh0, b_ih1, b_hh1,
                                                    packW, h0buf, h1buf, cnt, out);
  (void)in_sizes; (void)n_in; (void)out_size; (void)ws_size;
}
